// xLSTM_model_14903536517288
// MI455X (gfx1250) — matrix operand order repaired and checked by reading, not compiled, not run on hardware
//
#include <hip/hip_runtime.h>

// xLSTM forward for MI455X (gfx1250, wave32, WMMA).
// Round 3: single prep kernel (static-geometry weight packing + x f32->f16),
// pre-packed WMMA-B fragments, f16 row-major activations for A fragments,
// parallel epilogues. GEMMs: v_wmma_f32_16x16x32_f16 (f16 in, f32 accum).

#define NTOK 2048
#define HID  128
#define INW  64
#define LDSL 132  // padded LDS row stride (floats)

typedef __attribute__((ext_vector_type(16))) _Float16 v16h;
typedef __attribute__((ext_vector_type(8)))  _Float16 v8h;
typedef __attribute__((ext_vector_type(8)))  float    v8f;

__device__ __forceinline__ float sigm(float x) { return 1.0f / (1.0f + expf(-x)); }

// ---- fragment loaders ------------------------------------------------------
// A: 16x32 (MxK), f16 row-major source. Lane L: row M=L%16, half=L>>4.
// e=0..15 -> K = k0 + half*16 + e.
__device__ __forceinline__ v16h load_a16(const _Float16* __restrict__ Xh, int ldx,
                                         int row0, int k0, int lane) {
  const int m = lane & 15, half = lane >> 4;
  const _Float16* p = Xh + (row0 + m) * ldx + k0 + half * 16;
  return *(const v16h*)(p);
}

// B: pre-packed fragment order; tileIdx selects (col-tile, k-chunk).
__device__ __forceinline__ v16h load_b16(const _Float16* __restrict__ P,
                                         int tileIdx, int lane) {
  return *(const v16h*)(P + ((size_t)tileIdx * 32 + lane) * 16);
}

// ---- 4-matrix GEMM: Xh[row0:+16, 0:K] x {P0..P3} -> LDS (f32) --------------
// 8 waves/block, 32 tiles (4 mats x 8 col-tiles). DUAL accumulates a second
// packed operand set Q (slstm layer 1: pre = x@W + x@U, x == h_prev).
template <int K, bool DUAL>
__device__ __forceinline__ void gemm4(const _Float16* __restrict__ Xh, int ldx,
                                      const _Float16* __restrict__ P0, const _Float16* __restrict__ P1,
                                      const _Float16* __restrict__ P2, const _Float16* __restrict__ P3,
                                      const _Float16* __restrict__ Q0, const _Float16* __restrict__ Q1,
                                      const _Float16* __restrict__ Q2, const _Float16* __restrict__ Q3,
                                      float* lds, int row0) {
  constexpr int KC = K / 32;
  const int lane = threadIdx.x & 31;
  const int wave = threadIdx.x >> 5;
  const _Float16* Ps[4] = {P0, P1, P2, P3};
  const _Float16* Qs[4] = {Q0, Q1, Q2, Q3};
#pragma unroll
  for (int it = 0; it < 4; ++it) {
    const int tile = wave + it * 8;  // uniform per wave
    const int mat = tile >> 3;
    const int ct  = tile & 7;
    v8f acc = {};
#pragma unroll
    for (int kc = 0; kc < KC; ++kc) {
      v16h a = load_a16(Xh, ldx, row0, kc * 32, lane);
      v16h b = load_b16(Ps[mat], ct * KC + kc, lane);
      acc = __builtin_amdgcn_wmma_f32_16x16x32_f16(false, a, false, b, (short)0,
                                                   acc, false, false);
    }
    if (DUAL) {
#pragma unroll
      for (int kc = 0; kc < KC; ++kc) {
        v16h a = load_a16(Xh, ldx, row0, kc * 32, lane);
        v16h b = load_b16(Qs[mat], ct * KC + kc, lane);
        acc = __builtin_amdgcn_wmma_f32_16x16x32_f16(false, a, false, b, (short)0,
                                                     acc, false, false);
      }
    }
    // C/D layout: element r -> (M = r + 8*(lane>>4), N = lane&15)
    const int n = lane & 15, mh = lane >> 4;
    float* dst = lds + mat * (16 * LDSL);
#pragma unroll
    for (int r = 0; r < 8; ++r) dst[(mh * 8 + r) * LDSL + ct * 16 + n] = acc[r];
  }
}

// ---- Prep: pack all 20 weight matrices + convert x to f16 ------------------
// Packed blob layout (f16 elems): 4 blobs of 8192 (K=64), 16 blobs of 16384
// (K=128), total 294912; then ids >= 294912 convert x (131072 elems).
// Blob sizes are multiples of 256 => matrix index is uniform per block.
// dst[((ct*KC + kc)*32 + lane)*16 + e] =
//     W[(kc*32 + (lane>>4)*16 + e)*128 + ct*16 + (lane&15)]
struct PrepArgs { const float* w[20]; const float* x; };

__global__ __launch_bounds__(256) void k_prep(PrepArgs a,
                                              _Float16* __restrict__ WP,
                                              _Float16* __restrict__ xh) {
  const int id = blockIdx.x * blockDim.x + threadIdx.x;
  if (id < 32768) {                       // K=64 blobs (m0: q,k,v,o)
    const int mat = id >> 13, off = id & 8191;
    const int e = off & 15, lane = (off >> 4) & 31, rest = off >> 9;
    const int kc = rest & 1, ct = rest >> 1;  // KC = 2
    const int k = kc * 32 + (lane >> 4) * 16 + e;
    const int col = ct * 16 + (lane & 15);
    WP[id] = (_Float16)a.w[mat][k * HID + col];
  } else if (id < 294912) {               // K=128 blobs (m1 + s0 + s1)
    const int r = id - 32768;
    const int mat = 4 + (r >> 14), off = r & 16383;
    const int e = off & 15, lane = (off >> 4) & 31, rest = off >> 9;
    const int kc = rest & 3, ct = rest >> 2;  // KC = 4
    const int k = kc * 32 + (lane >> 4) * 16 + e;
    const int col = ct * 16 + (lane & 15);
    WP[id] = (_Float16)a.w[mat][k * HID + col];
  } else {                                // x f32 -> f16
    const int r = id - 294912;            // 0 .. 131071
    xh[r] = (_Float16)a.x[r];
  }
}

// ---- Kernel 1: mLSTM layer 0 (din=64, zero prior state) --------------------
// h0 = sig(o) * i0*v0*(k0.q0) / max(|i0*(k0.q0)|, 1)
__global__ __launch_bounds__(256) void k_mlstm0(
    const float* __restrict__ x, const _Float16* __restrict__ xh,
    const _Float16* __restrict__ pq, const _Float16* __restrict__ pk,
    const _Float16* __restrict__ pv, const _Float16* __restrict__ po,
    const float* __restrict__ bq, const float* __restrict__ bk,
    const float* __restrict__ bv, const float* __restrict__ bo,
    const float* __restrict__ wi, const float* __restrict__ bi,
    float* __restrict__ h0, _Float16* __restrict__ h0h,
    float* __restrict__ k0v, float* __restrict__ v0v, float* __restrict__ i0v) {
  __shared__ float lds[4 * 16 * LDSL];
  __shared__ float red[2][16][16];
  __shared__ float scal[1][16];
  const int row0 = blockIdx.x * 16;
  gemm4<INW, false>(xh, INW, pq, pk, pv, po, nullptr, nullptr, nullptr, nullptr,
                    lds, row0);
  __syncthreads();
  const int t = threadIdx.x >> 4, g = threadIdx.x & 15;
  const int tok = row0 + t;
  const float* lq = lds + 0 * 16 * LDSL + t * LDSL;
  const float* lk = lds + 1 * 16 * LDSL + t * LDSL;
  const float* lv = lds + 2 * 16 * LDSL + t * LDSL;
  const float* lo = lds + 3 * 16 * LDSL + t * LDSL;
  const float rs = 0.08838834764831845f;  // 1/sqrt(128)
  float pxi = 0.f;
#pragma unroll
  for (int u = 0; u < 4; ++u) pxi += x[tok * INW + g * 4 + u] * wi[g * 4 + u];
  float pd = 0.f;
#pragma unroll
  for (int u = 0; u < 8; ++u) {
    const int j = g * 8 + u;
    pd += ((lk[j] + bk[j]) * rs) * (lq[j] + bq[j]);
  }
  red[0][t][g] = pxi;
  red[1][t][g] = pd;
  __syncthreads();
  if (g == 0) {
    float sxi = bi[0], sd = 0.f;
    for (int u = 0; u < 16; ++u) { sxi += red[0][t][u]; sd += red[1][t][u]; }
    const float i0 = expf(sxi);
    const float denom = fmaxf(fabsf(i0 * sd), 1.0f);
    scal[0][t] = i0 * sd / denom;
    i0v[tok] = i0;
  }
  __syncthreads();
  const float scale = scal[0][t];
#pragma unroll
  for (int u = 0; u < 8; ++u) {
    const int j = g * 8 + u;
    const float kj = (lk[j] + bk[j]) * rs;
    const float vj = lv[j] + bv[j];
    const float h = sigm(lo[j] + bo[j]) * scale * vj;
    h0[tok * HID + j] = h;
    h0h[tok * HID + j] = (_Float16)h;
    k0v[tok * HID + j] = kj;
    v0v[tok * HID + j] = vj;
  }
}

// ---- Kernel 2: mLSTM layer 1 (din=128, state from layer 0) -----------------
__global__ __launch_bounds__(256) void k_mlstm1(
    const float* __restrict__ h0, const _Float16* __restrict__ h0h,
    const _Float16* __restrict__ pq, const _Float16* __restrict__ pk,
    const _Float16* __restrict__ pv, const _Float16* __restrict__ po,
    const float* __restrict__ bq, const float* __restrict__ bk,
    const float* __restrict__ bv, const float* __restrict__ bo,
    const float* __restrict__ wi, const float* __restrict__ bi,
    const float* __restrict__ wf, const float* __restrict__ bf,
    const float* __restrict__ k0v, const float* __restrict__ v0v,
    const float* __restrict__ i0v, _Float16* __restrict__ h1h) {
  __shared__ float lds[4 * 16 * LDSL];
  __shared__ float red[4][16][16];
  __shared__ float scal[2][16];
  const int row0 = blockIdx.x * 16;
  gemm4<HID, false>(h0h, HID, pq, pk, pv, po, nullptr, nullptr, nullptr, nullptr,
                    lds, row0);
  __syncthreads();
  const int t = threadIdx.x >> 4, g = threadIdx.x & 15;
  const int tok = row0 + t;
  const float* lq = lds + 0 * 16 * LDSL + t * LDSL;
  const float* lk = lds + 1 * 16 * LDSL + t * LDSL;
  const float* lv = lds + 2 * 16 * LDSL + t * LDSL;
  const float* lo = lds + 3 * 16 * LDSL + t * LDSL;
  const float rs = 0.08838834764831845f;
  float pxi = 0.f, pxf = 0.f, pd0 = 0.f, pd1 = 0.f;
#pragma unroll
  for (int u = 0; u < 8; ++u) {
    const int j = g * 8 + u;
    const float xj = h0[tok * HID + j];
    pxi += xj * wi[j];
    pxf += xj * wf[j];
    const float qj = lq[j] + bq[j];
    pd0 += k0v[tok * HID + j] * qj;
    pd1 += ((lk[j] + bk[j]) * rs) * qj;
  }
  red[0][t][g] = pxi; red[1][t][g] = pxf;
  red[2][t][g] = pd0; red[3][t][g] = pd1;
  __syncthreads();
  if (g == 0) {
    float sxi = bi[0], sxf = bf[0], sd0 = 0.f, sd1 = 0.f;
    for (int u = 0; u < 16; ++u) {
      sxi += red[0][t][u]; sxf += red[1][t][u];
      sd0 += red[2][t][u]; sd1 += red[3][t][u];
    }
    const float i1 = expf(sxi);
    const float f1 = sigm(sxf);
    const float i0 = i0v[tok];
    const float nq = f1 * i0 * sd0 + i1 * sd1;
    const float denom = fmaxf(fabsf(nq), 1.0f);
    scal[0][t] = f1 * i0 * sd0 / denom;  // multiplies v0
    scal[1][t] = i1 * sd1 / denom;       // multiplies v1
  }
  __syncthreads();
  const float s0c = scal[0][t], s1c = scal[1][t];
#pragma unroll
  for (int u = 0; u < 8; ++u) {
    const int j = g * 8 + u;
    const float vj = lv[j] + bv[j];
    const float h = sigm(lo[j] + bo[j]) * (s0c * v0v[tok * HID + j] + s1c * vj);
    h1h[tok * HID + j] = (_Float16)h;
  }
}

// ---- Kernel 3: sLSTM layer 0 (h_prev = 0, zero state) ----------------------
__global__ __launch_bounds__(256) void k_slstm0(
    const _Float16* __restrict__ yh,
    const _Float16* __restrict__ pi, const _Float16* __restrict__ pf,
    const _Float16* __restrict__ po, const _Float16* __restrict__ pc,
    const float* __restrict__ bi, const float* __restrict__ bf,
    const float* __restrict__ bo, const float* __restrict__ bc,
    _Float16* __restrict__ hoh, float* __restrict__ co,
    float* __restrict__ mo, float* __restrict__ no) {
  __shared__ float lds[4 * 16 * LDSL];
  const int row0 = blockIdx.x * 16;
  gemm4<HID, false>(yh, HID, pi, pf, po, pc, nullptr, nullptr, nullptr, nullptr,
                    lds, row0);
  __syncthreads();
#pragma unroll
  for (int e = 0; e < 8; ++e) {
    const int idx = threadIdx.x * 8 + e;  // 16 tok x 128
    const int t = idx >> 7, j = idx & 127;
    const int tok = row0 + t;
    const float gi = lds[0 * 16 * LDSL + t * LDSL + j] + bi[j];
    const float gf = lds[1 * 16 * LDSL + t * LDSL + j] + bf[j];
    const float go = lds[2 * 16 * LDSL + t * LDSL + j] + bo[j];
    const float gc = lds[3 * 16 * LDSL + t * LDSL + j] + bc[j];
    const float mn = fmaxf(gf, gi);  // m_prev = 0
    const float ip = expf(gi - mn);
    const float o  = sigm(go);
    const float z  = tanhf(gc);
    const float c  = ip * z;  // c_prev = 0
    const float n  = ip;      // n_prev = 0
    hoh[tok * HID + j] = (_Float16)(o * (c / n));
    co[tok * HID + j] = c;
    mo[tok * HID + j] = mn;
    no[tok * HID + j] = n;
  }
}

// ---- Kernel 4: sLSTM layer 1 (x == h_prev => pre = x@W + x@U + b) ----------
__global__ __launch_bounds__(256) void k_slstm1(
    const _Float16* __restrict__ yh,
    const _Float16* __restrict__ pwi, const _Float16* __restrict__ pwf,
    const _Float16* __restrict__ pwo, const _Float16* __restrict__ pwc,
    const _Float16* __restrict__ pui, const _Float16* __restrict__ puf,
    const _Float16* __restrict__ puo, const _Float16* __restrict__ puc,
    const float* __restrict__ bi, const float* __restrict__ bf,
    const float* __restrict__ bo, const float* __restrict__ bc,
    const float* __restrict__ ci, const float* __restrict__ mi,
    const float* __restrict__ ni, float* __restrict__ ho) {
  __shared__ float lds[4 * 16 * LDSL];
  const int row0 = blockIdx.x * 16;
  gemm4<HID, true>(yh, HID, pwi, pwf, pwo, pwc, pui, puf, puo, puc, lds, row0);
  __syncthreads();
#pragma unroll
  for (int e = 0; e < 8; ++e) {
    const int idx = threadIdx.x * 8 + e;
    const int t = idx >> 7, j = idx & 127;
    const int tok = row0 + t;
    const float gi = lds[0 * 16 * LDSL + t * LDSL + j] + bi[j];
    const float gf = lds[1 * 16 * LDSL + t * LDSL + j] + bf[j];
    const float go = lds[2 * 16 * LDSL + t * LDSL + j] + bo[j];
    const float gc = lds[3 * 16 * LDSL + t * LDSL + j] + bc[j];
    const float mn = fmaxf(gf + mi[tok * HID + j], gi);
    const float ip = expf(gi - mn);
    const float fp = expf(gf);  // as written in the reference
    const float o  = sigm(go);
    const float z  = tanhf(gc);
    const float c  = fp * ci[tok * HID + j] + ip * z;
    const float n  = fp * ni[tok * HID + j] + ip;
    ho[tok * HID + j] = o * (c / n);
  }
}

// ---- Kernel 5: final FC [2048,128] x [128,1] -------------------------------
__global__ __launch_bounds__(256) void k_fc(const float* __restrict__ y,
                                            const float* __restrict__ fw,
                                            const float* __restrict__ fb,
                                            float* __restrict__ out) {
  const int t = blockIdx.x * blockDim.x + threadIdx.x;
  if (t < NTOK) {
    float s = fb[0];
    const float4* yv = (const float4*)(y + t * HID);
    const float4* wv = (const float4*)fw;
#pragma unroll 8
    for (int j = 0; j < HID / 4; ++j) {
      const float4 a = yv[j], b = wv[j];
      s += a.x * b.x + a.y * b.y + a.z * b.z + a.w * b.w;
    }
    out[t] = s;
  }
}

// ---- Host launch -----------------------------------------------------------
// Input order = setup_inputs() insertion order:
//  0: x
//  m0: 1 wq 2 bq 3 wk 4 bk 5 wv 6 bv 7 wi 8 bi 9 wf 10 bf 11 wo 12 bo
//  m1: 13..24 (same layout)
//  s0: 25 wi 26 ui 27 bi 28 wf 29 uf 30 bf 31 wo 32 uo 33 bo 34 wc 35 uc 36 bc
//  s1: 37..48 (same layout)
//  49: fc_w  50: fc_b
extern "C" void kernel_launch(void* const* d_in, const int* in_sizes, int n_in,
                              void* d_out, int out_size, void* d_ws, size_t ws_size,
                              hipStream_t stream) {
  const float* in[64];
  for (int i = 0; i < n_in && i < 64; ++i) in[i] = (const float*)d_in[i];

  const size_t TB = (size_t)NTOK * HID;  // 262144
  float* B0 = (float*)d_ws;   // h0  -> later c
  float* B1 = B0 + TB;        // k0  -> later m
  float* B2 = B1 + TB;        // v0  -> later n
  float* B3 = B2 + TB;        // hs1 (fc input)
  float* I0 = B3 + TB;        // i0 scalars [2048]
  _Float16* F0 = (_Float16*)(I0 + NTOK);  // xh -> later h1h
  _Float16* F1 = F0 + TB;                 // h0h -> later hs0h
  _Float16* WP = F1 + TB;                 // packed weights (294912 f16)

  // Packed weight blob offsets (f16 elements): m0 4x8192, then 16x16384.
  _Float16* m0q = WP;            _Float16* m0k = WP + 8192;
  _Float16* m0v = WP + 16384;    _Float16* m0o = WP + 24576;
  _Float16* m1q = WP + 32768;    _Float16* m1k = WP + 49152;
  _Float16* m1v = WP + 65536;    _Float16* m1o = WP + 81920;
  _Float16* s0i = WP + 98304;    _Float16* s0f = WP + 114688;
  _Float16* s0o = WP + 131072;   _Float16* s0c = WP + 147456;
  _Float16* s1i = WP + 163840;   _Float16* s1f = WP + 180224;
  _Float16* s1o = WP + 196608;   _Float16* s1c = WP + 212992;
  _Float16* s1I = WP + 229376;   _Float16* s1F = WP + 245760;
  _Float16* s1O = WP + 262144;   _Float16* s1C = WP + 278528;

  dim3 blk(256);

  // -- stage 0: one prep launch: pack 20 weight mats + convert x --
  PrepArgs pa;
  pa.w[0] = in[1];  pa.w[1] = in[3];  pa.w[2] = in[5];  pa.w[3] = in[11];
  pa.w[4] = in[13]; pa.w[5] = in[15]; pa.w[6] = in[17]; pa.w[7] = in[23];
  pa.w[8] = in[25]; pa.w[9] = in[28]; pa.w[10] = in[31]; pa.w[11] = in[34];
  pa.w[12] = in[37]; pa.w[13] = in[40]; pa.w[14] = in[43]; pa.w[15] = in[46];
  pa.w[16] = in[38]; pa.w[17] = in[41]; pa.w[18] = in[44]; pa.w[19] = in[47];
  pa.x = in[0];
  k_prep<<<(294912 + NTOK * INW) / 256, blk, 0, stream>>>(pa, WP, F0);

  dim3 grid(NTOK / 16);
  // -- mLSTM layer 0: reads x/xh, writes h0(B0)+h0h(F1), k0(B1), v0(B2), i0 --
  k_mlstm0<<<grid, blk, 0, stream>>>(in[0], F0, m0q, m0k, m0v, m0o, in[2], in[4],
                                     in[6], in[12], in[7], in[8],
                                     B0, F1, B1, B2, I0);
  // -- mLSTM layer 1: reads h0/h0h + state, writes h1h(F0) --
  k_mlstm1<<<grid, blk, 0, stream>>>(B0, F1, m1q, m1k, m1v, m1o, in[14], in[16],
                                     in[18], in[24], in[19], in[20], in[21],
                                     in[22], B1, B2, I0, F0);
  // -- sLSTM layer 0: reads h1h(F0), writes hs0h(F1), c(B0), m(B1), n(B2) --
  k_slstm0<<<grid, blk, 0, stream>>>(F0, s0i, s0f, s0o, s0c, in[27], in[30],
                                     in[33], in[36], F1, B0, B1, B2);
  // -- sLSTM layer 1: reads hs0h(F1)+state, writes hs1(B3) --
  k_slstm1<<<grid, blk, 0, stream>>>(F1, s1i, s1f, s1o, s1c, s1I, s1F, s1O, s1C,
                                     in[39], in[42], in[45], in[48],
                                     B0, B1, B2, B3);
  // -- final FC --
  k_fc<<<NTOK / 256, blk, 0, stream>>>(B3, in[49], in[50], (float*)d_out);
}
